// coGN_36129264894718
// MI455X (gfx1250) — matrix operand order repaired and checked by reading, not compiled, not run on hardware
//
#include <hip/hip_runtime.h>
#include <hip/hip_bf16.h>

#define NN    20000
#define EE    640000
#define DD    128
#define KRBF  64
#define LLAY  3
#define GG    128
#define RCUT  5.0f
// delta = RCUT/KRBF = 5/64 ; 1/(2*delta^2) = 4096/50
#define INV2D2 81.92f
// mu spacing = RCUT/(KRBF-1)
#define MUSTEP 0.0793650793650794f

typedef __attribute__((ext_vector_type(8)))  __bf16 v8bf;
typedef __attribute__((ext_vector_type(16))) __bf16 v16bf;
typedef __attribute__((ext_vector_type(8)))  float  v8f;
typedef __attribute__((ext_vector_type(4)))  float  v4f;
typedef unsigned int u32x4 __attribute__((ext_vector_type(4)));
typedef int          i32x4 __attribute__((ext_vector_type(4)));
typedef int          i32x8 __attribute__((ext_vector_type(8)));

// fragment sizes (bf16 elements)
#define FRAG_MSG_ELEMS (10 * 8 * 512)   // 40960 -> 80 KB
#define FRAG_UPD_ELEMS (8 * 8 * 512)    // 32768 -> 64 KB

// ---- element property tables (module constants in the reference) ----
__device__ __constant__ float c_mass[119] = {
0.0f,1.008f,4.003f,6.941f,9.012f,10.81f,12.01f,14.01f,16.0f,19.0f,
20.18f,22.99f,24.31f,26.98f,28.09f,30.97f,32.07f,35.45f,39.95f,39.1f,
40.08f,44.96f,47.87f,50.94f,52.0f,54.94f,55.85f,58.93f,58.69f,63.55f,
65.38f,69.72f,72.63f,74.92f,78.97f,79.9f,83.8f,85.47f,87.62f,88.91f,
91.22f,92.91f,95.95f,98.0f,101.1f,102.9f,106.4f,107.9f,112.4f,114.8f,
118.7f,121.8f,127.6f,126.9f,131.3f,132.9f,137.3f,138.9f,140.1f,140.9f,
144.2f,145.0f,150.4f,152.0f,157.3f,158.9f,162.5f,164.9f,167.3f,168.9f,
173.0f,175.0f,178.5f,180.9f,183.8f,186.2f,190.2f,192.2f,195.1f,197.0f,
200.6f,204.4f,207.2f,209.0f,209.0f,210.0f,222.0f,223.0f,226.0f,227.0f,
232.0f,231.0f,238.0f,237.0f,244.0f,243.0f,247.0f,247.0f,251.0f,252.0f,
257.0f,258.0f,259.0f,262.0f,267.0f,270.0f,269.0f,270.0f,270.0f,278.0f,
281.0f,281.0f,285.0f,286.0f,289.0f,289.0f,293.0f,293.0f,294.0f};
__device__ __constant__ float c_rad[119] = {
0.0f,1.2f,1.4f,1.82f,1.53f,1.92f,1.7f,1.55f,1.52f,1.47f,
1.54f,2.27f,1.73f,1.84f,2.1f,1.8f,1.8f,1.75f,1.88f,2.75f,
2.31f,2.11f,2.0f,2.0f,2.0f,2.0f,2.0f,2.0f,1.63f,1.4f,
1.39f,1.87f,2.11f,1.85f,1.9f,1.85f,2.02f,3.03f,2.49f,2.0f,
2.0f,2.0f,2.0f,2.0f,2.0f,2.0f,1.63f,1.72f,1.58f,1.93f,
2.17f,2.06f,2.06f,1.98f,2.16f,3.43f,2.68f,2.0f,2.0f,2.0f,
2.0f,2.0f,2.0f,2.0f,2.0f,2.0f,2.0f,2.0f,2.0f,2.0f,
2.0f,2.0f,2.0f,2.0f,2.0f,2.0f,2.0f,2.0f,1.75f,1.66f,
1.55f,1.96f,2.02f,2.07f,1.97f,2.02f,2.2f,3.48f,2.83f,2.0f,
2.0f,2.0f,1.86f,2.0f,2.0f,2.0f,2.0f,2.0f,2.0f,2.0f,
2.0f,2.0f,2.0f,2.0f,2.0f,2.0f,2.0f,2.0f,2.0f,2.0f,
2.0f,2.0f,2.0f,2.0f,2.0f,2.0f,2.0f,2.0f,2.0f};
__device__ __constant__ float c_en[119] = {
0.0f,2.2f,0.0f,0.98f,1.57f,2.04f,2.55f,3.04f,3.44f,3.98f,
0.0f,0.93f,1.31f,1.61f,1.9f,2.19f,2.58f,3.16f,0.0f,0.82f,
1.0f,1.36f,1.54f,1.63f,1.66f,1.55f,1.83f,1.88f,1.91f,1.9f,
1.65f,1.81f,2.01f,2.18f,2.55f,2.96f,3.0f,0.82f,0.95f,1.22f,
1.33f,1.6f,2.16f,1.9f,2.2f,2.28f,2.2f,1.93f,1.69f,1.78f,
1.96f,2.05f,2.1f,2.66f,2.6f,0.79f,0.89f,1.1f,1.12f,1.13f,
1.14f,1.13f,1.17f,1.2f,1.2f,1.22f,1.23f,1.24f,1.25f,1.1f,
1.27f,1.3f,1.5f,2.36f,1.9f,2.2f,2.2f,2.28f,2.54f,2.0f,
1.62f,1.87f,2.33f,2.02f,2.0f,2.2f,2.2f,0.7f,0.9f,1.1f,
1.3f,1.5f,1.38f,1.36f,1.28f,1.3f,1.3f,1.3f,1.3f,1.3f,
1.3f,1.3f,1.3f,1.3f,1.3f,1.3f,1.3f,1.3f,1.3f,1.3f,
1.3f,1.3f,1.3f,1.3f,1.3f,1.3f,1.3f,1.3f,1.3f};
__device__ __constant__ float c_ion[119] = {
0.0f,13.6f,24.59f,5.39f,9.32f,8.3f,11.26f,14.53f,13.62f,17.42f,
21.56f,5.14f,7.65f,5.99f,8.15f,10.49f,10.36f,12.97f,15.76f,4.34f,
6.11f,6.56f,6.83f,6.75f,6.77f,7.43f,7.9f,7.88f,7.64f,7.73f,
9.39f,6.0f,7.9f,9.79f,9.75f,11.81f,14.0f,4.18f,5.69f,6.22f,
6.63f,6.76f,7.09f,7.28f,7.36f,7.46f,8.34f,7.58f,8.99f,5.79f,
7.34f,8.64f,9.01f,10.45f,12.13f,3.89f,5.21f,5.58f,5.54f,5.47f,
5.53f,5.58f,5.64f,5.67f,6.15f,5.86f,5.94f,6.02f,6.11f,6.18f,
6.25f,5.43f,6.83f,7.55f,7.86f,7.83f,8.44f,8.97f,8.96f,9.23f,
10.44f,6.11f,7.42f,7.29f,8.42f,9.3f,10.75f,4.07f,5.28f,5.17f,
6.31f,5.89f,6.19f,6.27f,6.03f,5.97f,6.02f,6.2f,6.28f,6.42f,
6.5f,6.58f,6.65f,4.9f,6.0f,6.0f,6.0f,6.0f,6.0f,6.0f,
6.0f,6.0f,6.0f,6.0f,6.0f,6.0f,6.0f,6.0f,6.0f};

__device__ __forceinline__ void atomic_add_f32(float* p, float v) {
  __hip_atomic_fetch_add(p, v, __ATOMIC_RELAXED, __HIP_MEMORY_SCOPE_AGENT);
}
__device__ __forceinline__ v16bf cat8(v8bf lo, v8bf hi) {
  return __builtin_shufflevector(lo, hi, 0,1,2,3,4,5,6,7,8,9,10,11,12,13,14,15);
}

// ---- stage a flat block of global memory into LDS (offset 0) via the TDM ----
// nbytes must be a multiple of 8. ldsOff = byte offset of destination in LDS.
__device__ __forceinline__ void stage_to_lds(const void* gsrc, void* lds_generic,
                                             unsigned ldsOff, unsigned nbytes,
                                             int tid, int nthreads) {
#if __has_builtin(__builtin_amdgcn_tensor_load_to_lds)
  if (tid < 32) {
    const unsigned long long ga = (unsigned long long)gsrc;
    const unsigned units = nbytes >> 3;       // 8-byte elements
    u32x4 g0;
    g0[0] = 1u;                               // count = 1, user descriptor
    g0[1] = ldsOff;                           // lds_addr
    g0[2] = (unsigned)ga;                     // global_addr[31:0]
    g0[3] = (unsigned)((ga >> 32) & 0x01FFFFFFu) | (2u << 30); // addr[56:32] | type=2
    i32x8 g1;
    g1[0] = (3 << 16);                        // data_size = 8B, no multicast
    g1[1] = (int)((units & 0xFFFFu) << 16);   // tensor_dim0[15:0] in bits 47:32? -> [79:48] low
    g1[2] = (int)((units >> 16) & 0xFFFFu) | (1 << 16); // tensor_dim0 hi | tensor_dim1 = 1
    g1[3] = (int)((units & 0xFFFFu) << 16);   // tile_dim0 = units
    g1[4] = 0;                                // tile_dim1/2 unused
    g1[5] = (int)units;                       // tensor_dim0_stride
    g1[6] = 0;
    g1[7] = 0;
    i32x4 gz4 = {0, 0, 0, 0};
#if defined(__clang_major__) && (__clang_major__ >= 23)
    i32x8 gz8 = {0, 0, 0, 0, 0, 0, 0, 0};
    __builtin_amdgcn_tensor_load_to_lds(g0, g1, gz4, gz4, gz8, 0);
#else
    __builtin_amdgcn_tensor_load_to_lds(g0, g1, gz4, gz4, 0);
#endif
#if __has_builtin(__builtin_amdgcn_s_wait_tensorcnt)
    __builtin_amdgcn_s_wait_tensorcnt(0);
#endif
  }
#else
  const uint4* g = (const uint4*)gsrc;
  uint4* l = (uint4*)lds_generic;
  const int nchunks = (int)(nbytes >> 4);
  for (int i = tid; i < nchunks; i += nthreads) l[i] = g[i];
#endif
  __syncthreads();
}

// ---------------- weight pre-swizzle to WMMA B-fragment order ----------------
// Per fragment (512 bf16): low-half elements (e=0..7) of all 32 lanes packed as
// lane*8+j (512 B), then high-half (e=8..15) at +256 elements. This makes each
// ds_load_b128 read consecutive 16B per lane -> bank-conflict-free.
// value(lane, e) = w[l][kc*32 + (e>>3)*16 + (lane>>4)*8 + (e&7)][nt*16 + (lane&15)]
__global__ void k_prep_frag(const float* __restrict__ w, __bf16* __restrict__ frag,
                            int kdim, int total) {
  int i = blockIdx.x * blockDim.x + threadIdx.x;
  if (i >= total) return;
  int f    = i & 511;                 // offset within fragment
  int rest2= i >> 9;                  // (l*KC + kc)*8 + nt
  int nt   = rest2 & 7;
  int rest = rest2 >> 3;              // l*KC + kc
  int KC   = kdim >> 5;
  int l    = rest / KC;
  int kc   = rest - l * KC;
  int hs   = f >> 8;                  // 0 = elements 0..7, 1 = elements 8..15
  int lane = (f >> 3) & 31;
  int j0   = f & 7;
  int e    = hs * 8 + j0;
  int k    = kc * 32 + hs * 16 + (lane >> 4) * 8 + j0;
  int n    = nt * 16 + (lane & 15);
  frag[i] = (__bf16)w[((size_t)l * kdim + k) * DD + n];
}

// ---------------- node init: h0 = [emb(z)|props(z)] @ proj_w + b ----------------
__global__ void __launch_bounds__(128) k_node_init(
    const int* __restrict__ z, const float* __restrict__ emb,
    const float* __restrict__ pw, const float* __restrict__ pb,
    float* __restrict__ h, __bf16* __restrict__ hb) {
  __shared__ float s_in[DD + 4];
  int n = blockIdx.x, t = threadIdx.x;
  int zz = z[n]; zz = zz < 0 ? 0 : (zz > 118 ? 118 : zz);
  s_in[t] = emb[(size_t)zz * DD + t];
  if (t == 0) {
    s_in[DD + 0] = c_mass[zz]; s_in[DD + 1] = c_rad[zz];
    s_in[DD + 2] = c_en[zz];   s_in[DD + 3] = c_ion[zz];
  }
  __syncthreads();
  float acc = pb[t];
  #pragma unroll 4
  for (int k = 0; k < DD + 4; ++k) acc += s_in[k] * pw[(size_t)k * DD + t];
  h [(size_t)n * DD + t] = acc;
  hb[(size_t)n * DD + t] = (__bf16)acc;
}

__global__ void k_zero(float* __restrict__ p, long n) {
  long i = (long)blockIdx.x * blockDim.x + threadIdx.x;
  if (i < n) p[i] = 0.0f;
}

// ---------------- edge kernel: msg = relu([h_s|h_d|rbf] @ Wmsg + b); scatter ----------------
// Dynamic LDS: [ frag 80KB | bias 512B | dst cache 512B ]
__global__ void __launch_bounds__(256) k_edge_msg(
    const int* __restrict__ src, const int* __restrict__ dst,
    const float* __restrict__ edist, const __bf16* __restrict__ hb,
    const __bf16* __restrict__ fragw,   // layer-offset: 10 kc x 8 nt x 512
    const float* __restrict__ bias,     // layer-offset: 128
    float* __restrict__ agg) {
  extern __shared__ char smem[];
  __bf16* s_frag = (__bf16*)smem;
  float*  s_bias = (float*)(smem + FRAG_MSG_ELEMS * 2);
  int*    s_dst  = (int*)  (smem + FRAG_MSG_ELEMS * 2 + 512);

  const int tid  = threadIdx.x;
  const int wv   = tid >> 5;
  const int lane = tid & 31;
  const int row  = lane & 15;
  const int half = lane >> 4;
  const int col  = row;

  if (tid < DD) s_bias[tid] = bias[tid];
  stage_to_lds(fragw, s_frag, 0u, FRAG_MSG_ELEMS * 2, tid, blockDim.x);

  const int NT     = EE / 16;
  const int gw     = blockIdx.x * 8 + wv;
  const int nwaves = gridDim.x * 8;

  for (int tile = gw; tile < NT; tile += nwaves) {
    const long base = (long)tile * 16;
    const int   s    = src[base + row];
    const int   d    = dst[base + row];
    const float dist = edist[base + row];
    if (lane < 16) s_dst[wv * 16 + row] = d;

    v8f acc[8];
    #pragma unroll
    for (int nt = 0; nt < 8; ++nt) {
      const float bv = s_bias[nt * 16 + col];
      #pragma unroll
      for (int r = 0; r < 8; ++r) acc[nt][r] = bv;
    }

    auto mm8 = [&](v16bf A, int kc) {
      const __bf16* fb = s_frag + (size_t)kc * 8 * 512;
      #pragma unroll
      for (int nt = 0; nt < 8; ++nt) {
        v8bf blo = *(const v8bf*)(fb + (size_t)nt * 512 + lane * 8);
        v8bf bhi = *(const v8bf*)(fb + (size_t)nt * 512 + 256 + lane * 8);
        acc[nt] = __builtin_amdgcn_wmma_f32_16x16x32_bf16(
            false, A, false, cat8(blo, bhi), (short)0, acc[nt], false, false);
      }
    };

    // K-chunks 0..3 : h[src] ; 4..7 : h[dst]
    const __bf16* rs = hb + (size_t)s * DD;
    const __bf16* rd = hb + (size_t)d * DD;
    #pragma unroll
    for (int kc = 0; kc < 8; ++kc) {
      const __bf16* rp = (kc < 4) ? rs : rd;
      const int kl = (kc & 3) * 32 + half * 8;
      v8bf lo = *(const v8bf*)(rp + kl);
      v8bf hi = *(const v8bf*)(rp + kl + 16);
      mm8(cat8(lo, hi), kc);
    }
    // K-chunks 8..9 : RBF computed on the fly
    #pragma unroll
    for (int kc = 8; kc < 10; ++kc) {
      v16bf A;
      #pragma unroll
      for (int j = 0; j < 8; ++j) {
        const int k0 = (kc - 8) * 32 + half * 8 + j;
        const float t0 = dist - (float)k0 * MUSTEP;
        const float t1 = dist - (float)(k0 + 16) * MUSTEP;
        A[j]     = (__bf16)__expf(-t0 * t0 * INV2D2);
        A[8 + j] = (__bf16)__expf(-t1 * t1 * INV2D2);
      }
      mm8(A, kc);
    }

    // relu + scatter-add into agg[dst]
    #pragma unroll
    for (int nt = 0; nt < 8; ++nt) {
      #pragma unroll
      for (int r = 0; r < 8; ++r) {
        float v = acc[nt][r];
        v = v > 0.0f ? v : 0.0f;
        const int nd = s_dst[wv * 16 + half * 8 + r];
        atomic_add_f32(&agg[(size_t)nd * DD + nt * 16 + col], v);
      }
    }
  }
}

// ---------------- node update: h += relu([h|agg] @ Wupd + b) ----------------
// Dynamic LDS: [ frag 64KB | bias 512B ]
__global__ void __launch_bounds__(256) k_node_upd(
    const float* __restrict__ agg,
    const __bf16* __restrict__ fragw,   // layer-offset: 8 kc x 8 nt x 512
    const float* __restrict__ bias,
    float* __restrict__ h, __bf16* __restrict__ hb, int ntiles) {
  extern __shared__ char smem[];
  __bf16* s_frag = (__bf16*)smem;
  float*  s_bias = (float*)(smem + FRAG_UPD_ELEMS * 2);

  const int tid  = threadIdx.x;
  const int wv   = tid >> 5;
  const int lane = tid & 31;
  const int row  = lane & 15;
  const int half = lane >> 4;
  const int col  = row;

  if (tid < DD) s_bias[tid] = bias[tid];
  stage_to_lds(fragw, s_frag, 0u, FRAG_UPD_ELEMS * 2, tid, blockDim.x);

  const int gw     = blockIdx.x * 8 + wv;
  const int nwaves = gridDim.x * 8;

  for (int tile = gw; tile < ntiles; tile += nwaves) {
    const long nb = (long)tile * 16;

    v8f acc[8];
    #pragma unroll
    for (int nt = 0; nt < 8; ++nt) {
      const float bv = s_bias[nt * 16 + col];
      #pragma unroll
      for (int r = 0; r < 8; ++r) acc[nt][r] = bv;
    }

    auto mm8 = [&](v16bf A, int kc) {
      const __bf16* fb = s_frag + (size_t)kc * 8 * 512;
      #pragma unroll
      for (int nt = 0; nt < 8; ++nt) {
        v8bf blo = *(const v8bf*)(fb + (size_t)nt * 512 + lane * 8);
        v8bf bhi = *(const v8bf*)(fb + (size_t)nt * 512 + 256 + lane * 8);
        acc[nt] = __builtin_amdgcn_wmma_f32_16x16x32_bf16(
            false, A, false, cat8(blo, bhi), (short)0, acc[nt], false, false);
      }
    };

    // K-chunks 0..3 : h (bf16 mirror)
    const __bf16* rh = hb + (size_t)(nb + row) * DD;
    #pragma unroll
    for (int kc = 0; kc < 4; ++kc) {
      const int kl = kc * 32 + half * 8;
      v8bf lo = *(const v8bf*)(rh + kl);
      v8bf hi = *(const v8bf*)(rh + kl + 16);
      mm8(cat8(lo, hi), kc);
    }
    // K-chunks 4..7 : agg (fp32 -> bf16 on the fly)
    const float* ra = agg + (size_t)(nb + row) * DD;
    #pragma unroll
    for (int kc = 4; kc < 8; ++kc) {
      const int kl = (kc - 4) * 32 + half * 8;
      v4f f0 = *(const v4f*)(ra + kl);
      v4f f1 = *(const v4f*)(ra + kl + 4);
      v4f f2 = *(const v4f*)(ra + kl + 16);
      v4f f3 = *(const v4f*)(ra + kl + 20);
      v16bf A;
      #pragma unroll
      for (int j = 0; j < 4; ++j) {
        A[j]      = (__bf16)f0[j];
        A[4 + j]  = (__bf16)f1[j];
        A[8 + j]  = (__bf16)f2[j];
        A[12 + j] = (__bf16)f3[j];
      }
      mm8(A, kc);
    }

    // h_new = h + relu(upd); refresh bf16 mirror
    #pragma unroll
    for (int nt = 0; nt < 8; ++nt) {
      #pragma unroll
      for (int r = 0; r < 8; ++r) {
        float u = acc[nt][r];
        u = u > 0.0f ? u : 0.0f;
        const size_t idx = (size_t)(nb + half * 8 + r) * DD + nt * 16 + col;
        const float nh = h[idx] + u;
        h[idx]  = nh;
        hb[idx] = (__bf16)nh;
      }
    }
  }
}

// ---------------- pooling + readout ----------------
__global__ void k_pool_accum(const float* __restrict__ h, const int* __restrict__ bid,
                             float* __restrict__ pooled, float* __restrict__ cnts) {
  long i = (long)blockIdx.x * blockDim.x + threadIdx.x;
  if (i >= (long)NN * DD) return;
  const int node = (int)(i >> 7);
  const int f    = (int)(i & (DD - 1));
  const int g    = bid[node];
  atomic_add_f32(&pooled[(size_t)g * DD + f], h[i]);
  if (f == 0) atomic_add_f32(&cnts[g], 1.0f);
}

__global__ void __launch_bounds__(128) k_out(
    const float* __restrict__ pooled, const float* __restrict__ cnts,
    const float* __restrict__ w_out, const float* __restrict__ b_out,
    float* __restrict__ out) {
  __shared__ float sred[DD];
  const int g = blockIdx.x, t = threadIdx.x;
  float c = cnts[g];
  c = c > 1.0f ? c : 1.0f;
  sred[t] = pooled[(size_t)g * DD + t] / c * w_out[t];
  __syncthreads();
  #pragma unroll
  for (int s = DD / 2; s > 0; s >>= 1) {
    if (t < s) sred[t] += sred[t + s];
    __syncthreads();
  }
  if (t == 0) out[g] = sred[0] + b_out[0];
}

// ---------------- host orchestration ----------------
extern "C" void kernel_launch(void* const* d_in, const int* in_sizes, int n_in,
                              void* d_out, int out_size, void* d_ws, size_t ws_size,
                              hipStream_t stream) {
  const int*   z     = (const int*)  d_in[0];
  const int*   src   = (const int*)  d_in[1];
  const int*   dst   = (const int*)  d_in[2];
  const int*   bid   = (const int*)  d_in[3];
  const float* edist = (const float*)d_in[4];
  const float* emb   = (const float*)d_in[5];
  const float* pw    = (const float*)d_in[6];
  const float* pb    = (const float*)d_in[7];
  const float* wmsg  = (const float*)d_in[8];
  const float* bmsg  = (const float*)d_in[9];
  const float* wupd  = (const float*)d_in[10];
  const float* bupd  = (const float*)d_in[11];
  const float* wout  = (const float*)d_in[12];
  const float* bout  = (const float*)d_in[13];
  float* out = (float*)d_out;

  char* ws = (char*)d_ws;
  size_t off = 0;
  auto carve = [&](size_t bytes) -> char* {
    char* p = ws + off;
    off = (off + bytes + 255) & ~(size_t)255;
    return p;
  };
  float*  h      = (float*) carve((size_t)NN * DD * 4);
  __bf16* hb     = (__bf16*)carve((size_t)NN * DD * 2);
  float*  agg    = (float*) carve((size_t)NN * DD * 4);
  __bf16* fmsg   = (__bf16*)carve((size_t)LLAY * FRAG_MSG_ELEMS * 2);
  __bf16* fupd   = (__bf16*)carve((size_t)LLAY * FRAG_UPD_ELEMS * 2);
  float*  pooled = (float*) carve((size_t)GG * DD * 4);
  float*  cnts   = (float*) carve((size_t)GG * 4);

  const int nfm = LLAY * FRAG_MSG_ELEMS;
  const int nfu = LLAY * FRAG_UPD_ELEMS;
  k_prep_frag<<<(nfm + 255) / 256, 256, 0, stream>>>(wmsg, fmsg, 2 * DD + KRBF, nfm);
  k_prep_frag<<<(nfu + 255) / 256, 256, 0, stream>>>(wupd, fupd, 2 * DD,        nfu);
  k_node_init<<<NN, 128, 0, stream>>>(z, emb, pw, pb, h, hb);

  const int ntiles = NN / 16;
  const size_t smem_edge = (size_t)FRAG_MSG_ELEMS * 2 + 512 + 512;
  const size_t smem_upd  = (size_t)FRAG_UPD_ELEMS * 2 + 512;
  for (int l = 0; l < LLAY; ++l) {
    k_zero<<<((long)NN * DD + 255) / 256, 256, 0, stream>>>(agg, (long)NN * DD);
    k_edge_msg<<<500, 256, smem_edge, stream>>>(src, dst, edist, hb,
        fmsg + (size_t)l * FRAG_MSG_ELEMS, bmsg + l * DD, agg);
    k_node_upd<<<20, 256, smem_upd, stream>>>(agg,
        fupd + (size_t)l * FRAG_UPD_ELEMS, bupd + l * DD, h, hb, ntiles);
  }

  k_zero<<<((long)GG * DD + 255) / 256, 256, 0, stream>>>(pooled, (long)GG * DD);
  k_zero<<<1, 256, 0, stream>>>(cnts, (long)GG);
  k_pool_accum<<<((long)NN * DD + 255) / 256, 256, 0, stream>>>(h, bid, pooled, cnts);
  k_out<<<GG, 128, 0, stream>>>(pooled, cnts, wout, bout, out);
}
